// EllipseRoIHeads_77292231459158
// MI455X (gfx1250) — matrix operand order repaired and checked by reading, not compiled, not run on hardware
//
#include <hip/hip_runtime.h>

// ---------------- problem constants (match reference) ----------------
#define B_IMG 2
#define N_PROP 2000
#define N_GT 16
#define NPA (N_PROP + N_GT)          // 2016 proposals+gt per image
#define C_FEAT 256
#define H_FEAT 128
#define W_FEAT 128
#define NUM_CLASSES 2
#define BPI 512
#define NUM_POS_MAX 128              // BPI * 0.25
#define POOL 7
#define D_IN (C_FEAT * POOL * POOL)  // 12544
#define HID 1024
#define NROI (B_IMG * BPI)           // 1024
#define FSCALE (1.0f / 8.0f)
#define SBETA (1.0f / 9.0f)

typedef __attribute__((ext_vector_type(16))) _Float16 v16h;
typedef __attribute__((ext_vector_type(8)))  _Float16 v8h;
typedef __attribute__((ext_vector_type(8)))  float    v8f;

union V16U { v16h v; v8h h[2]; };

__device__ __forceinline__ unsigned pack2h(float a, float b) {
    union { unsigned u; _Float16 h[2]; } p;
    p.h[0] = (_Float16)a;
    p.h[1] = (_Float16)b;
    return p.u;
}

// ---------------- JAX threefry2x32 (key = (0,42)) ----------------
__device__ __forceinline__ unsigned rotl32(unsigned x, int r) {
    return (x << r) | (x >> (32 - r));
}
__device__ __forceinline__ void threefry2x32(unsigned ks0, unsigned ks1,
                                             unsigned x0, unsigned x1,
                                             unsigned& o0, unsigned& o1) {
    unsigned ks2 = ks0 ^ ks1 ^ 0x1BD11BDAu;
    x0 += ks0; x1 += ks1;
    const int ra[4] = {13, 15, 26, 6};
    const int rb[4] = {17, 29, 16, 24};
    #pragma unroll
    for (int i = 0; i < 4; ++i) { x0 += x1; x1 = rotl32(x1, ra[i]); x1 ^= x0; }
    x0 += ks1; x1 += ks2 + 1u;
    #pragma unroll
    for (int i = 0; i < 4; ++i) { x0 += x1; x1 = rotl32(x1, rb[i]); x1 ^= x0; }
    x0 += ks2; x1 += ks0 + 2u;
    #pragma unroll
    for (int i = 0; i < 4; ++i) { x0 += x1; x1 = rotl32(x1, ra[i]); x1 ^= x0; }
    x0 += ks0; x1 += ks1 + 3u;
    #pragma unroll
    for (int i = 0; i < 4; ++i) { x0 += x1; x1 = rotl32(x1, rb[i]); x1 ^= x0; }
    x0 += ks1; x1 += ks2 + 4u;
    #pragma unroll
    for (int i = 0; i < 4; ++i) { x0 += x1; x1 = rotl32(x1, ra[i]); x1 ^= x0; }
    x0 += ks2; x1 += ks0 + 5u;
    o0 = x0; o1 = x1;
}
__device__ __forceinline__ float bits_to_uniform(unsigned bits) {
    unsigned u = (bits >> 9) | 0x3f800000u;
    return __uint_as_float(u) - 1.0f;
}

// ---------------- kernel 0: zero loss accumulators ----------------
__global__ void init_kernel(float* acc) {
    acc[0] = 0.0f;
    acc[1] = 0.0f;
}

// ---------------- kernel 1: IoU matching + labels ----------------
__global__ void match_kernel(const float* __restrict__ proposals,
                             const float* __restrict__ gt_boxes,
                             const int* __restrict__ gt_labels,
                             int* __restrict__ matches,
                             int* __restrict__ labs) {
    int g = blockIdx.x * blockDim.x + threadIdx.x;
    if (g >= B_IMG * NPA) return;
    int b = g / NPA;
    int j = g % NPA;
    float bx0, bx1, bx2, bx3;
    if (j < N_PROP) {
        const float* p = proposals + ((size_t)b * N_PROP + j) * 4;
        bx0 = p[0]; bx1 = p[1]; bx2 = p[2]; bx3 = p[3];
    } else {
        const float* p = gt_boxes + ((size_t)b * N_GT + (j - N_PROP)) * 4;
        bx0 = p[0]; bx1 = p[1]; bx2 = p[2]; bx3 = p[3];
    }
    float a1 = (bx2 - bx0) * (bx3 - bx1);
    float best = -1.0f;
    int bi = 0;
    for (int t = 0; t < N_GT; ++t) {
        const float* q = gt_boxes + ((size_t)b * N_GT + t) * 4;
        float a2 = (q[2] - q[0]) * (q[3] - q[1]);
        float lx = fmaxf(bx0, q[0]), ly = fmaxf(bx1, q[1]);
        float rx = fminf(bx2, q[2]), ry = fminf(bx3, q[3]);
        float iw = fmaxf(rx - lx, 0.0f), ih = fmaxf(ry - ly, 0.0f);
        float inter = iw * ih;
        float iou = inter / (a1 + a2 - inter);
        if (iou > best) { best = iou; bi = t; }
    }
    matches[g] = bi;
    int lab = gt_labels[b * N_GT + bi];
    labs[g] = (best < 0.5f) ? 0 : lab;
}

// ---------------- kernel 2: sampling + gather + target encoding ----------------
__global__ __launch_bounds__(256)
void sample_kernel(const float* __restrict__ proposals,
                   const float* __restrict__ gt_boxes,
                   const float* __restrict__ gt_ell,
                   const int* __restrict__ matches,
                   const int* __restrict__ labs_in,
                   float* __restrict__ boxes,
                   int* __restrict__ labels,
                   float* __restrict__ targets) {
    __shared__ float rnd[NPA];
    __shared__ float prio[NPA];
    __shared__ int lab_s[NPA];
    int b = blockIdx.x;
    int tid = threadIdx.x;
    // rand matching jax.random.uniform(key(42), (B, NPA)): counter pair (i, i+NPA)
    for (int i = tid; i < NPA; i += 256) {
        unsigned o0, o1;
        threefry2x32(0u, 42u, (unsigned)i, (unsigned)(i + NPA), o0, o1);
        rnd[i] = bits_to_uniform(b == 0 ? o0 : o1);
        lab_s[i] = labs_in[b * NPA + i];
    }
    __syncthreads();
    // rank positives by rand (descending, index tie-break), cap, build priorities
    for (int i = tid; i < NPA; i += 256) {
        float p;
        if (lab_s[i] > 0) {
            float ri = rnd[i];
            int cnt = 0;
            for (int j = 0; j < NPA; ++j) {
                if (lab_s[j] > 0) {
                    float rj = rnd[j];
                    if (rj > ri || (rj == ri && j < i)) cnt++;
                }
            }
            p = (cnt < NUM_POS_MAX) ? (ri + 2.0f) : -1.0e9f;
        } else {
            p = rnd[i];
        }
        prio[i] = p;
    }
    __syncthreads();
    // top-BPI by priority (stable); emit gathered boxes/labels/targets
    for (int i = tid; i < NPA; i += 256) {
        float pi = prio[i];
        int cnt = 0;
        for (int j = 0; j < NPA; ++j) {
            float pj = prio[j];
            if (pj > pi || (pj == pi && j < i)) cnt++;
        }
        if (cnt < BPI) {
            int s = b * BPI + cnt;
            float bx0, bx1, bx2, bx3;
            if (i < N_PROP) {
                const float* p = proposals + ((size_t)b * N_PROP + i) * 4;
                bx0 = p[0]; bx1 = p[1]; bx2 = p[2]; bx3 = p[3];
            } else {
                const float* p = gt_boxes + ((size_t)b * N_GT + (i - N_PROP)) * 4;
                bx0 = p[0]; bx1 = p[1]; bx2 = p[2]; bx3 = p[3];
            }
            boxes[s * 4 + 0] = bx0; boxes[s * 4 + 1] = bx1;
            boxes[s * 4 + 2] = bx2; boxes[s * 4 + 3] = bx3;
            labels[s] = lab_s[i];
            int m = matches[b * NPA + i];
            const float* e = gt_ell + ((size_t)b * N_GT + m) * 5;
            float ea = e[0], eb = e[1], ex = e[2], ey = e[3], th = e[4];
            float w = fmaxf(bx2 - bx0, 1.0f);
            float h = fmaxf(bx3 - bx1, 1.0f);
            float cx = 0.5f * (bx0 + bx2);
            float cy = 0.5f * (bx1 + bx3);
            targets[s * 6 + 0] = (ex - cx) / w;
            targets[s * 6 + 1] = (ey - cy) / h;
            targets[s * 6 + 2] = logf(fmaxf(2.0f * ea, 0.001f) / w);
            targets[s * 6 + 3] = logf(fmaxf(2.0f * eb, 0.001f) / h);
            targets[s * 6 + 4] = sinf(2.0f * th);
            targets[s * 6 + 5] = cosf(2.0f * th);
        }
    }
}

// ---------------- kernel 3: RoI align -> f16 GEMM operand ----------------
__global__ __launch_bounds__(256)
void roi_align_kernel(const float* __restrict__ features,
                      const float* __restrict__ boxes,
                      _Float16* __restrict__ xh) {
    int bin = blockIdx.x;          // 0..48
    int r   = blockIdx.y;          // 0..NROI-1
    int c   = threadIdx.x;         // 0..255
    int py = bin / POOL, px = bin % POOL;
    int img = r / BPI;
    const float* bx = boxes + (size_t)r * 4;
    float x1 = bx[0] * FSCALE, y1 = bx[1] * FSCALE;
    float x2 = bx[2] * FSCALE, y2 = bx[3] * FSCALE;
    float bw = fmaxf(x2 - x1, 1.0f) / (float)POOL;
    float bh = fmaxf(y2 - y1, 1.0f) / (float)POOL;
    const float* f = features + ((size_t)img * C_FEAT + c) * (H_FEAT * W_FEAT);
    float acc = 0.0f;
    #pragma unroll
    for (int s = 0; s < 4; ++s) {
        int sy = s >> 1, sx = s & 1;
        float yy = y1 + ((float)(py * 2 + sy) + 0.5f) * 0.5f * bh;
        float xx = x1 + ((float)(px * 2 + sx) + 0.5f) * 0.5f * bw;
        yy = fminf(fmaxf(yy, 0.0f), (float)(H_FEAT - 1));
        xx = fminf(fmaxf(xx, 0.0f), (float)(W_FEAT - 1));
        int y0 = (int)floorf(yy); y0 = y0 < 0 ? 0 : (y0 > H_FEAT - 2 ? H_FEAT - 2 : y0);
        int x0 = (int)floorf(xx); x0 = x0 < 0 ? 0 : (x0 > W_FEAT - 2 ? W_FEAT - 2 : x0);
        float ly = yy - (float)y0, lx = xx - (float)x0;
        const float* p = f + y0 * W_FEAT + x0;
        float v00 = p[0], v01 = p[1], v10 = p[W_FEAT], v11 = p[W_FEAT + 1];
        acc += v00 * (1.0f - ly) * (1.0f - lx) + v01 * (1.0f - ly) * lx +
               v10 * ly * (1.0f - lx) + v11 * ly * lx;
    }
    xh[(size_t)r * D_IN + c * (POOL * POOL) + bin] = (_Float16)(acc * 0.25f);
}

// ---------------- kernel 4: pipelined tiled WMMA GEMM, fused bias+ReLU ----------------
// C[M,N] = relu(A[M,K] (f16, row-major) * B[K,N] (f32, row-major, cast to f16) + bias)
// BM=BN=128, BK=64; 8 waves, each owns 64x32 of C (4x2 WMMA tiles).
// Register-staged pipeline: stage tile k -> LDS, barrier, issue global loads for
// tile k+1, run 16 WMMAs for tile k while loads are in flight.
template <bool OUT_HALF>
__global__ __launch_bounds__(256)
void gemm_wmma(const _Float16* __restrict__ A,
               const float* __restrict__ B,
               const float* __restrict__ bias,
               void* __restrict__ Out,
               int M, int N, int K) {
    constexpr int BM = 128, BN = 128, BK = 64;
    constexpr int LDA = BK + 8;   // 72 halfs = 144B rows (16B aligned)
    constexpr int LDB = BK + 8;   // Bs stored transposed: [BN][BK]
    __shared__ _Float16 As[BM * LDA];
    __shared__ _Float16 Bs[BN * LDB];

    const int tid  = threadIdx.x;
    const int lane = tid & 31;
    const int wave = tid >> 5;    // 0..7
    const int wm = wave & 1;      // M half (64 rows)
    const int wn = wave >> 1;     // N quarter (32 cols)
    const int m0 = blockIdx.x * BM;
    const int n0 = blockIdx.y * BN;

    const int lrow = lane & 15;
    const int ha  = (lane & 16) ? 8  : 0;   // A per-lane K pattern (interleaved)
    const int kb2 = (lane & 16) ? 8 : 0;    // B per-lane K pattern (interleaved)

    // staging thread mapping
    const int ar = tid >> 1;            // A: row 0..127
    const int ac = (tid & 1) * 32;      // A: 32 halfs (64B) per thread
    const int bk = (tid >> 4) * 4;      // B: 4 consecutive K rows
    const int bn = (tid & 15) * 8;      // B: 8 consecutive N cols

    v8f zero = {0.f, 0.f, 0.f, 0.f, 0.f, 0.f, 0.f, 0.f};
    v8f acc[4][2];
    #pragma unroll
    for (int mi = 0; mi < 4; ++mi)
        #pragma unroll
        for (int ni = 0; ni < 2; ++ni) acc[mi][ni] = zero;

    v8h  a_reg[4];
    float b_reg[4][8];

    auto load_tiles = [&](int k0) {
        const _Float16* asrc = A + (size_t)(m0 + ar) * K + k0 + ac;
        a_reg[0] = *(const v8h*)(asrc);
        a_reg[1] = *(const v8h*)(asrc + 8);
        a_reg[2] = *(const v8h*)(asrc + 16);
        a_reg[3] = *(const v8h*)(asrc + 24);
        const float* bsrc = B + (size_t)(k0 + bk) * N + n0 + bn;
        #pragma unroll
        for (int r = 0; r < 4; ++r)
            #pragma unroll
            for (int j = 0; j < 8; ++j)
                b_reg[r][j] = bsrc[(size_t)r * N + j];
        if (k0 + BK < K) {
            __builtin_prefetch(asrc + BK, 0, 1);
            __builtin_prefetch(bsrc + (size_t)BK * N, 0, 1);
        }
    };
    auto store_tiles = [&]() {
        _Float16* adst = &As[ar * LDA + ac];
        *(v8h*)(adst)      = a_reg[0];
        *(v8h*)(adst + 8)  = a_reg[1];
        *(v8h*)(adst + 16) = a_reg[2];
        *(v8h*)(adst + 24) = a_reg[3];
        #pragma unroll
        for (int j = 0; j < 8; ++j) {
            *(unsigned*)&Bs[(bn + j) * LDB + bk]     = pack2h(b_reg[0][j], b_reg[1][j]);
            *(unsigned*)&Bs[(bn + j) * LDB + bk + 2] = pack2h(b_reg[2][j], b_reg[3][j]);
        }
    };

    load_tiles(0);
    for (int k0 = 0; k0 < K; k0 += BK) {
        store_tiles();
        __syncthreads();
        if (k0 + BK < K) load_tiles(k0 + BK);   // overlap with WMMA below

        #pragma unroll
        for (int kc = 0; kc < BK; kc += 32) {
            #pragma unroll
            for (int mi = 0; mi < 4; ++mi) {
                const _Float16* ap = &As[(wm * 64 + mi * 16 + lrow) * LDA + kc + ha];
                V16U a;
                a.h[0] = *(const v8h*)(ap);        // K = kc+ha .. +7
                a.h[1] = *(const v8h*)(ap + 16);   // K = kc+16+ha .. +7
                #pragma unroll
                for (int ni = 0; ni < 2; ++ni) {
                    const _Float16* bp = &Bs[(wn * 32 + ni * 16 + lrow) * LDB + kc + kb2];
                    V16U bf;
                    bf.h[0] = *(const v8h*)(bp);      // K = kc+kb2 .. +7
                    bf.h[1] = *(const v8h*)(bp + 16); // K = kc+16+kb2 .. +7
                    acc[mi][ni] = __builtin_amdgcn_wmma_f32_16x16x32_f16(
                        false, a.v, false, bf.v, (short)0, acc[mi][ni], false, false);
                }
            }
        }
        __syncthreads();
    }

    // epilogue: D layout — VGPR r: lanes 0-15 -> M=r, lanes 16-31 -> M=8+r; N=lane%16
    #pragma unroll
    for (int mi = 0; mi < 4; ++mi) {
        #pragma unroll
        for (int ni = 0; ni < 2; ++ni) {
            int col = n0 + wn * 32 + ni * 16 + lrow;
            float bv = bias[col];
            #pragma unroll
            for (int r = 0; r < 8; ++r) {
                int row = m0 + wm * 64 + mi * 16 + r + ((lane & 16) ? 8 : 0);
                float v = acc[mi][ni][r] + bv;
                v = v > 0.0f ? v : 0.0f;
                if (OUT_HALF)
                    ((_Float16*)Out)[(size_t)row * N + col] = (_Float16)v;
                else
                    ((float*)Out)[(size_t)row * N + col] = v;
            }
        }
    }
}

// ---------------- kernel 5: heads + losses ----------------
__global__ __launch_bounds__(256)
void heads_loss_kernel(const float* __restrict__ x2,
                       const float* __restrict__ wc, const float* __restrict__ bc,
                       const float* __restrict__ wr, const float* __restrict__ br,
                       const int* __restrict__ labels,
                       const float* __restrict__ targets,
                       float* __restrict__ acc) {
    __shared__ float red[256][16];
    int n = blockIdx.x;
    int tid = threadIdx.x;
    float o[14];
    #pragma unroll
    for (int j = 0; j < 14; ++j) o[j] = 0.0f;
    for (int k = tid; k < HID; k += 256) {
        float v = x2[(size_t)n * HID + k];
        o[0] += v * wc[k * NUM_CLASSES + 0];
        o[1] += v * wc[k * NUM_CLASSES + 1];
        #pragma unroll
        for (int j = 0; j < 12; ++j) o[2 + j] += v * wr[k * 12 + j];
    }
    #pragma unroll
    for (int j = 0; j < 14; ++j) red[tid][j] = o[j];
    __syncthreads();
    for (int s = 128; s > 0; s >>= 1) {
        if (tid < s) {
            #pragma unroll
            for (int j = 0; j < 14; ++j) red[tid][j] += red[tid + s][j];
        }
        __syncthreads();
    }
    if (tid == 0) {
        float l0 = red[0][0] + bc[0];
        float l1 = red[0][1] + bc[1];
        float mx = fmaxf(l0, l1);
        float lse = mx + logf(expf(l0 - mx) + expf(l1 - mx));
        int lab = labels[n];
        float ce = lse - (lab == 0 ? l0 : l1);
        atomicAdd(&acc[0], ce);
        if (lab > 0) {
            float rs = 0.0f;
            #pragma unroll
            for (int j = 0; j < 6; ++j) {
                float p = red[0][2 + lab * 6 + j] + br[lab * 6 + j];
                float d = fabsf(p - targets[n * 6 + j]);
                rs += (d < SBETA) ? 0.5f * d * d / SBETA : d - 0.5f * SBETA;
            }
            atomicAdd(&acc[1], rs);
        }
    }
}

__global__ void finalize_kernel(const float* __restrict__ acc, float* __restrict__ out) {
    out[0] = acc[0] / (float)NROI;
    out[1] = acc[1] / (float)NROI;
}

// ---------------- host launcher ----------------
extern "C" void kernel_launch(void* const* d_in, const int* in_sizes, int n_in,
                              void* d_out, int out_size, void* d_ws, size_t ws_size,
                              hipStream_t stream) {
    const float* features    = (const float*)d_in[0];
    const float* proposals   = (const float*)d_in[1];
    const float* gt_boxes    = (const float*)d_in[2];
    const int*   gt_labels   = (const int*)  d_in[3];
    const float* gt_ellipses = (const float*)d_in[4];
    const float* w1 = (const float*)d_in[5];
    const float* b1 = (const float*)d_in[6];
    const float* w2 = (const float*)d_in[7];
    const float* b2 = (const float*)d_in[8];
    const float* wc = (const float*)d_in[9];
    const float* bc = (const float*)d_in[10];
    const float* wr = (const float*)d_in[11];
    const float* br = (const float*)d_in[12];

    char* w = (char*)d_ws;
    size_t off = 0;
    auto take = [&](size_t bytes) -> void* {
        void* p = w + off;
        off = (off + bytes + 255) & ~(size_t)255;
        return p;
    };
    _Float16* xh  = (_Float16*)take((size_t)NROI * D_IN * sizeof(_Float16)); // 25.7 MB
    _Float16* x1h = (_Float16*)take((size_t)NROI * HID * sizeof(_Float16));  // 2 MB
    float*    x2  = (float*)   take((size_t)NROI * HID * sizeof(float));     // 4 MB
    int*      matches = (int*) take((size_t)B_IMG * NPA * sizeof(int));
    int*      labs    = (int*) take((size_t)B_IMG * NPA * sizeof(int));
    float*    boxes   = (float*)take((size_t)NROI * 4 * sizeof(float));
    int*      labels  = (int*) take((size_t)NROI * sizeof(int));
    float*    targets = (float*)take((size_t)NROI * 6 * sizeof(float));
    float*    acc     = (float*)take(256);
    (void)ws_size; (void)n_in; (void)in_sizes; (void)out_size;

    init_kernel<<<1, 1, 0, stream>>>(acc);

    int nmatch = B_IMG * NPA;
    match_kernel<<<(nmatch + 255) / 256, 256, 0, stream>>>(
        proposals, gt_boxes, gt_labels, matches, labs);

    sample_kernel<<<B_IMG, 256, 0, stream>>>(
        proposals, gt_boxes, gt_ellipses, matches, labs, boxes, labels, targets);

    roi_align_kernel<<<dim3(POOL * POOL, NROI), 256, 0, stream>>>(features, boxes, xh);

    gemm_wmma<true><<<dim3(NROI / 128, HID / 128), 256, 0, stream>>>(
        xh, w1, b1, (void*)x1h, NROI, HID, D_IN);

    gemm_wmma<false><<<dim3(NROI / 128, HID / 128), 256, 0, stream>>>(
        x1h, w2, b2, (void*)x2, NROI, HID, HID);

    heads_loss_kernel<<<NROI, 256, 0, stream>>>(
        x2, wc, bc, wr, br, labels, targets, acc);

    finalize_kernel<<<1, 1, 0, stream>>>(acc, (float*)d_out);
}
